// FusedMoEModularKernel_84215718740362
// MI455X (gfx1250) — matrix operand order repaired and checked by reading, not compiled, not run on hardware
//
#include <hip/hip_runtime.h>
#include <hip/hip_bf16.h>
#include <stdint.h>

// ---------------- problem constants ----------------
#define M_TOK 4096
#define K_DIM 2048
#define N_DIM 5632
#define E_NUM 8
#define TOPK  2

// ---------------- tiling ----------------
#define BM      128           // token-slot tile (rows per block)
#define BN_ACT  64            // act columns per block (gate+up pair)
#define BK      32            // bf16 reduction step == one WMMA K
#define BK2     64            // GEMM2 output columns per block
#define LDS_STRIDE (BK + 8)   // 40 halfs = 80B row; keeps 16B align, skews banks

#define SLOT_CAP  9216        // 72*128 >= M*TOPK + E*(BM-1) = 9208
#define MAX_TILES (SLOT_CAP / BM)   // 72

typedef __attribute__((ext_vector_type(8)))  float   v8f;
typedef __attribute__((ext_vector_type(16))) __bf16  v16bf;
typedef __attribute__((ext_vector_type(8)))  __bf16  v8bf;

// ---------------- helpers ----------------
// bf16 = high 16 bits of fp32 after +0x8000 rounding; pack 2 per v_perm_b32.
__device__ __forceinline__ uint32_t pack_bf2(float x, float y) {
  union { float f; uint32_t u; } a, b;
  a.f = x; b.f = y;
  uint32_t ua = a.u + 0x8000u;
  uint32_t ub = b.u + 0x8000u;
  // result = (bf(y) << 16) | bf(x): bytes {ub[3],ub[2],ua[3],ua[2]}
  return __builtin_amdgcn_perm(ub, ua, 0x07060302u);
}
__device__ __forceinline__ uint16_t f2bf_u16(float f) {
  union { float f; uint32_t u; } in; in.f = f;
  return (uint16_t)((in.u + 0x8000u) >> 16);
}
__device__ __forceinline__ void store_bf4(__bf16* p, float4 v) {
  uint2 pk;
  pk.x = pack_bf2(v.x, v.y);
  pk.y = pack_bf2(v.z, v.w);
  *(uint2*)p = pk;            // 8B-aligned ds_store_b64
}
__device__ __forceinline__ v16bf mk16(v8bf lo, v8bf hi) {
  v16bf r;
#pragma unroll
  for (int i = 0; i < 8; i++) { r[i] = lo[i]; r[i + 8] = hi[i]; }
  return r;
}
__device__ __forceinline__ v8f v8f_zero() {
  v8f z;
#pragma unroll
  for (int i = 0; i < 8; i++) z[i] = 0.0f;
  return z;
}

// ---------------- kernel 0: zero output ----------------
__global__ void zero_out_kernel(float* __restrict__ out, int n) {
  int stride = gridDim.x * blockDim.x;
  for (int i = blockIdx.x * blockDim.x + threadIdx.x; i < n; i += stride)
    out[i] = 0.0f;
}

// ---------------- kernel 1: routing ----------------
// g_off[0..8]: BM-padded per-expert slot offsets; g_off[9]: padded total.
// g_rows[slot]: source token row (-1 = padding); g_rw[slot]: routing weight.
__global__ void moe_route(const int* __restrict__ topk_ids,
                          const float* __restrict__ topk_w,
                          int* __restrict__ g_off,
                          int* __restrict__ g_rows,
                          float* __restrict__ g_rw) {
  __shared__ int s_cnt[E_NUM];
  __shared__ int s_off[E_NUM + 1];
  __shared__ int s_cur[E_NUM];
  const int tid = threadIdx.x;
  if (tid < E_NUM) { s_cnt[tid] = 0; s_cur[tid] = 0; }
  __syncthreads();
  for (int i = tid; i < M_TOK * TOPK; i += 256)
    atomicAdd(&s_cnt[topk_ids[i]], 1);
  __syncthreads();
  if (tid == 0) {
    int acc = 0;
    for (int e = 0; e < E_NUM; e++) {
      s_off[e] = acc;
      acc += ((s_cnt[e] + BM - 1) / BM) * BM;
    }
    s_off[E_NUM] = acc;
    for (int e = 0; e <= E_NUM; e++) g_off[e] = s_off[e];
    g_off[9] = acc;   // padded total
  }
  __syncthreads();
  // init all slots as padding first
  for (int i = tid; i < SLOT_CAP; i += 256) { g_rows[i] = -1; g_rw[i] = 0.0f; }
  __syncthreads();
  // scatter routed entries
  for (int i = tid; i < M_TOK * TOPK; i += 256) {
    int e = topk_ids[i];
    int pos = s_off[e] + atomicAdd(&s_cur[e], 1);
    g_rows[pos] = i / TOPK;
    g_rw[pos]   = topk_w[i];
  }
}

// ---------------- kernel 2: GEMM1 (a1 @ w1e^T) + silu*up -> act (bf16) ----
// Double-buffered LDS, one barrier per K-stage; global loads for stage i+1
// overlap the 8 WMMAs of stage i.
__global__ __launch_bounds__(256)
void moe_gemm1(const float* __restrict__ a1, const float* __restrict__ w1,
               const int* __restrict__ g_off, const int* __restrict__ g_rows,
               unsigned short* __restrict__ g_act) {
  __shared__ int s_meta[10];
  __shared__ int s_rows[BM];
  __shared__ alignas(16) __bf16 As[2][BM][LDS_STRIDE];
  __shared__ alignas(16) __bf16 Bs[2][2 * BN_ACT][LDS_STRIDE]; // 0-63 gate, 64-127 up

  const int tid = threadIdx.x;
  const int slot0 = blockIdx.y * BM;
  if (tid < 10) s_meta[tid] = g_off[tid];
  for (int i = tid; i < BM; i += 256) s_rows[i] = g_rows[slot0 + i];
  __syncthreads();
  if (slot0 >= s_meta[9]) return;           // beyond padded total (uniform)
  int e = 0;
  while (e < E_NUM - 1 && slot0 >= s_meta[e + 1]) e++;

  const int n0 = blockIdx.x * BN_ACT;       // act column base
  const float* __restrict__ w1e = w1 + (size_t)e * (2 * N_DIM) * K_DIM;

  const int wid = tid >> 5, lane = tid & 31;
  const int wm = wid & 3;                   // wave row-tile (32 rows)
  const int wn = wid >> 2;                  // wave col-tile (32 act cols)
  const int lr = lane & 15;
  const int khalf = lane >> 4;              // 0/1: K-half per WMMA layout

  // ---- per-thread staging addresses, hoisted out of the K loop ----
  // Padding slots (row<0) are clamped to row 0: their act rows are computed
  // from garbage but never consumed (GEMM2 epilogue drops row<0 slots).
  const float* aptr[4]; const float* bptr[4];
  int sr[4], sc[4];
#pragma unroll
  for (int t = 0; t < 4; t++) {
    int idx = tid + 256 * t;                // [0,1024): 128 rows x 8 float4
    int r = idx >> 3, c4 = (idx & 7) * 4;
    sr[t] = r; sc[t] = c4;
    int arow = s_rows[r];
    if (arow < 0) arow = 0;
    aptr[t] = a1 + (size_t)arow * K_DIM + c4;
    int wr = (r < BN_ACT) ? (n0 + r) : (N_DIM + n0 + (r - BN_ACT));
    bptr[t] = w1e + (size_t)wr * K_DIM + c4;
  }

  v8f accG[2][2], accU[2][2];
#pragma unroll
  for (int a = 0; a < 2; a++)
#pragma unroll
    for (int b = 0; b < 2; b++) { accG[a][b] = v8f_zero(); accU[a][b] = v8f_zero(); }

  // ---- prologue: stage 0 -> buffer 0 ----
  float4 aR[4], bR[4];
#pragma unroll
  for (int t = 0; t < 4; t++) {
    aR[t] = *(const float4*)aptr[t];
    bR[t] = *(const float4*)bptr[t];
  }
#pragma unroll
  for (int t = 0; t < 4; t++) {
    store_bf4(&As[0][sr[t]][sc[t]], aR[t]);
    store_bf4(&Bs[0][sr[t]][sc[t]], bR[t]);
  }
  __syncthreads();

  const int NIT = K_DIM / BK;               // 64 stages
  for (int it = 0; it < NIT; ++it) {
    const int pb = it & 1;
    // issue global loads for stage it+1 (latency hidden by WMMAs below)
    if (it + 1 < NIT) {
      const int k0 = (it + 1) * BK;
#pragma unroll
      for (int t = 0; t < 4; t++) {
        aR[t] = *(const float4*)(aptr[t] + k0);
        bR[t] = *(const float4*)(bptr[t] + k0);
        __builtin_prefetch(bptr[t] + k0 + BK, 0, 0);
      }
    }

    // ---- compute stage it from buffer pb ----
    v16bf afr[2], bfrG[2], bfrU[2];
#pragma unroll
    for (int mt = 0; mt < 2; mt++) {
      // A 16x32: lanes 0-15 row M=l, K 0-15; lanes 16-31 K 16-31
      const __bf16* p = &As[pb][wm * 32 + mt * 16 + lr][khalf * 16];
      afr[mt] = mk16(*(const v8bf*)p, *(const v8bf*)(p + 8));
    }
#pragma unroll
    for (int nt = 0; nt < 2; nt++) {
      // B 32x16: lanes 0-15 col N=l, K 0-15; lanes 16-31 K 16-31
      const __bf16* pg = &Bs[pb][wn * 32 + nt * 16 + lr][khalf * 16];
      bfrG[nt] = mk16(*(const v8bf*)pg, *(const v8bf*)(pg + 8));
      const __bf16* pu = &Bs[pb][BN_ACT + wn * 32 + nt * 16 + lr][khalf * 16];
      bfrU[nt] = mk16(*(const v8bf*)pu, *(const v8bf*)(pu + 8));
    }
#pragma unroll
    for (int mt = 0; mt < 2; mt++)
#pragma unroll
      for (int nt = 0; nt < 2; nt++) {
        accG[mt][nt] = __builtin_amdgcn_wmma_f32_16x16x32_bf16(
            false, afr[mt], false, bfrG[nt], (short)0, accG[mt][nt], false, false);
        accU[mt][nt] = __builtin_amdgcn_wmma_f32_16x16x32_bf16(
            false, afr[mt], false, bfrU[nt], (short)0, accU[mt][nt], false, false);
      }

    // ---- store stage it+1 into the other buffer; single barrier per stage ----
    if (it + 1 < NIT) {
#pragma unroll
      for (int t = 0; t < 4; t++) {
        store_bf4(&As[pb ^ 1][sr[t]][sc[t]], aR[t]);
        store_bf4(&Bs[pb ^ 1][sr[t]][sc[t]], bR[t]);
      }
      __syncthreads();
    }
  }

  // ---- epilogue: act = silu(gate) * up, store bf16 ----
  const int mrow0 = slot0 + wm * 32;
#pragma unroll
  for (int mt = 0; mt < 2; mt++)
#pragma unroll
    for (int nt = 0; nt < 2; nt++)
#pragma unroll
      for (int i = 0; i < 8; i++) {
        float g = accG[mt][nt][i];
        float u = accU[mt][nt][i];
        float act = (g / (1.0f + __expf(-g))) * u;
        // D layout: VGPR i -> M = i + 8*(lane>=16); N = lane&15
        int m = mt * 16 + khalf * 8 + i;
        int n = nt * 16 + lr;
        g_act[(size_t)(mrow0 + m) * N_DIM + (n0 + wn * 32 + n)] = f2bf_u16(act);
      }
}

// ---------------- kernel 3: GEMM2 (act @ w2e^T), weighted atomic scatter ----
__global__ __launch_bounds__(256)
void moe_gemm2(const float* __restrict__ w2, const int* __restrict__ g_off,
               const int* __restrict__ g_rows, const float* __restrict__ g_rw,
               const unsigned short* __restrict__ g_act,
               float* __restrict__ out) {
  __shared__ int   s_meta[10];
  __shared__ int   s_rows[BM];
  __shared__ float s_rw[BM];
  __shared__ alignas(16) __bf16 As[2][BM][LDS_STRIDE];
  __shared__ alignas(16) __bf16 Bs[2][BK2][LDS_STRIDE];

  const int tid = threadIdx.x;
  const int slot0 = blockIdx.y * BM;
  if (tid < 10) s_meta[tid] = g_off[tid];
  for (int i = tid; i < BM; i += 256) {
    s_rows[i] = g_rows[slot0 + i];
    s_rw[i]   = g_rw[slot0 + i];
  }
  __syncthreads();
  if (slot0 >= s_meta[9]) return;
  int e = 0;
  while (e < E_NUM - 1 && slot0 >= s_meta[e + 1]) e++;

  const int kcol0 = blockIdx.x * BK2;       // output column base (over K_DIM)
  const float* __restrict__ w2e = w2 + (size_t)e * K_DIM * N_DIM;

  const int wid = tid >> 5, lane = tid & 31;
  const int wm = wid & 3, wn = wid >> 2;
  const int lr = lane & 15;
  const int khalf = lane >> 4;

  // hoisted staging addresses
  const unsigned short* aptr[2]; const float* bptr[2];
  int ar[2], ac[2], br[2], bc[2];
#pragma unroll
  for (int t = 0; t < 2; t++) {
    int idx = tid + 256 * t;                // [0,512)
    int r = idx >> 2, c = (idx & 3) * 8;    // act: 128 rows x 4x(8 halfs)
    ar[t] = r; ac[t] = c;
    aptr[t] = g_act + (size_t)(slot0 + r) * N_DIM + c;
    int r2 = idx >> 3, c4 = (idx & 7) * 4;  // w2: 64 rows x 8 float4
    br[t] = r2; bc[t] = c4;
    bptr[t] = w2e + (size_t)(kcol0 + r2) * N_DIM + c4;
  }

  v8f acc[2][2];
#pragma unroll
  for (int a = 0; a < 2; a++)
#pragma unroll
    for (int b = 0; b < 2; b++) acc[a][b] = v8f_zero();

  // ---- prologue: stage 0 -> buffer 0 ----
  uint4 aRu[2]; float4 bR[2];
#pragma unroll
  for (int t = 0; t < 2; t++) {
    aRu[t] = *(const uint4*)aptr[t];
    bR[t]  = *(const float4*)bptr[t];
  }
#pragma unroll
  for (int t = 0; t < 2; t++) {
    *(uint4*)(&As[0][ar[t]][ac[t]]) = aRu[t];       // act already bf16
    store_bf4(&Bs[0][br[t]][bc[t]], bR[t]);
  }
  __syncthreads();

  const int NIT = N_DIM / BK;               // 176 stages
  for (int it = 0; it < NIT; ++it) {
    const int pb = it & 1;
    if (it + 1 < NIT) {
      const int nb = (it + 1) * BK;
#pragma unroll
      for (int t = 0; t < 2; t++) {
        aRu[t] = *(const uint4*)(aptr[t] + nb);
        bR[t]  = *(const float4*)(bptr[t] + nb);
        __builtin_prefetch(bptr[t] + nb + BK, 0, 0);
      }
    }

    v16bf afr[2], bfr[2];
#pragma unroll
    for (int mt = 0; mt < 2; mt++) {
      const __bf16* p = &As[pb][wm * 32 + mt * 16 + lr][khalf * 16];
      afr[mt] = mk16(*(const v8bf*)p, *(const v8bf*)(p + 8));
    }
#pragma unroll
    for (int nt = 0; nt < 2; nt++) {
      const __bf16* p = &Bs[pb][wn * 32 + nt * 16 + lr][khalf * 16];
      bfr[nt] = mk16(*(const v8bf*)p, *(const v8bf*)(p + 8));
    }
#pragma unroll
    for (int mt = 0; mt < 2; mt++)
#pragma unroll
      for (int nt = 0; nt < 2; nt++)
        acc[mt][nt] = __builtin_amdgcn_wmma_f32_16x16x32_bf16(
            false, afr[mt], false, bfr[nt], (short)0, acc[mt][nt], false, false);

    if (it + 1 < NIT) {
#pragma unroll
      for (int t = 0; t < 2; t++) {
        *(uint4*)(&As[pb ^ 1][ar[t]][ac[t]]) = aRu[t];
        store_bf4(&Bs[pb ^ 1][br[t]][bc[t]], bR[t]);
      }
      __syncthreads();
    }
  }

  // ---- epilogue: out[row, k] += w * y  (<=2 contributions per element) ----
#pragma unroll
  for (int mt = 0; mt < 2; mt++)
#pragma unroll
    for (int nt = 0; nt < 2; nt++)
#pragma unroll
      for (int i = 0; i < 8; i++) {
        int m = mt * 16 + khalf * 8 + i;
        int lslot = wm * 32 + m;
        int row = s_rows[lslot];
        if (row >= 0) {
          float w = s_rw[lslot];
          int kc = kcol0 + wn * 32 + nt * 16 + lr;
          unsafeAtomicAdd(out + (size_t)row * K_DIM + kc, w * acc[mt][nt][i]);
        }
      }
}

// ---------------- launch ----------------
extern "C" void kernel_launch(void* const* d_in, const int* in_sizes, int n_in,
                              void* d_out, int out_size, void* d_ws, size_t ws_size,
                              hipStream_t stream) {
  const float* a1  = (const float*)d_in[0];
  const float* w1  = (const float*)d_in[1];
  const float* w2  = (const float*)d_in[2];
  const float* tkw = (const float*)d_in[3];
  const int*   tki = (const int*)d_in[4];
  float* out = (float*)d_out;
  char* ws = (char*)d_ws;

  // workspace layout
  int*   g_off  = (int*)ws;                              // 16 ints
  int*   g_rows = (int*)(ws + 64);                       // SLOT_CAP ints
  float* g_rw   = (float*)(ws + 64 + (size_t)SLOT_CAP * 4);
  size_t act_off = ((64 + (size_t)SLOT_CAP * 8) + 255) & ~(size_t)255;
  unsigned short* g_act = (unsigned short*)(ws + act_off);
  // total ws need: act_off + SLOT_CAP*N_DIM*2 ~= 99 MB

  zero_out_kernel<<<2048, 256, 0, stream>>>(out, M_TOK * K_DIM);
  moe_route<<<1, 256, 0, stream>>>(tki, tkw, g_off, g_rows, g_rw);
  moe_gemm1<<<dim3(N_DIM / BN_ACT, MAX_TILES), 256, 0, stream>>>(
      a1, w1, g_off, g_rows, g_act);
  moe_gemm2<<<dim3(K_DIM / BK2, MAX_TILES), 256, 0, stream>>>(
      w2, g_off, g_rows, g_rw, g_act, out);
}
